// LaPool_68934225101286
// MI455X (gfx1250) — compile-verified, matrix operand order checked by reading, not run on hardware
//
#include <hip/hip_runtime.h>

#define BB 32
#define NN 512
#define FF 128
#define KK 64
#define EPSC 1e-8f
#define SIGMA 0.8f
#define BN_EPS 1e-5f

typedef _Float16 half_t;
typedef __attribute__((ext_vector_type(8)))  _Float16 v8h;
typedef __attribute__((ext_vector_type(16))) _Float16 v16h;
typedef __attribute__((ext_vector_type(8)))  float    v8f;

// ---------------------------------------------------------------- utilities
__global__ void k_zero_f32(float* p, long n) {
  long i = (long)blockIdx.x * blockDim.x + threadIdx.x;
  if (i < n) p[i] = 0.f;
}

__global__ void k_f32_to_f16(const float* __restrict__ s, half_t* __restrict__ d, long n) {
  long i = (long)blockIdx.x * blockDim.x + threadIdx.x;
  if (i < n) d[i] = (half_t)s[i];
}

// Mh = (f16)(A + I) per graph  (bakes the GIN self-loop into the operand)
__global__ void k_a_to_m_f16(const float* __restrict__ s, half_t* __restrict__ d, long n) {
  long i = (long)blockIdx.x * blockDim.x + threadIdx.x;
  if (i >= n) return;
  long rem = i % ((long)NN * NN);
  int r = (int)(rem / NN), c = (int)(rem % NN);
  d[i] = (half_t)(s[i] + (r == c ? 1.f : 0.f));
}

// d[b][c][r] = (f16) s[b][r][c]
__global__ void k_transpose_to_f16(const float* __restrict__ s, half_t* __restrict__ d,
                                   int R, int C, long n) {
  long i = (long)blockIdx.x * blockDim.x + threadIdx.x;
  if (i >= n) return;
  long rc = (long)R * C;
  long b = i / rc;
  long rem = i - b * rc;
  int r = (int)(rem / C), c = (int)(rem % C);
  d[b * rc + (long)c * R + r] = (half_t)s[i];
}

__global__ void k_copy_f32(const float* __restrict__ s, float* __restrict__ d, long n) {
  long i = (long)blockIdx.x * blockDim.x + threadIdx.x;
  if (i < n) d[i] = s[i];
}

// ------------------------------------------------- adjacency build (scatter)
__global__ void k_scatter_edges(const long long* __restrict__ ei, float* __restrict__ A, long E) {
  long e = (long)blockIdx.x * blockDim.x + threadIdx.x;
  if (e >= E) return;
  long s = ei[e];
  long t = ei[E + e];
  long b = s / NN;
  atomicAdd(&A[b * (long)NN * NN + (s % NN) * (long)NN + (t % NN)], 1.f);
}

__global__ void k_deg_nd(const float* __restrict__ A, float* __restrict__ deg,
                         float* __restrict__ inv_nd) {
  int i = blockIdx.x * blockDim.x + threadIdx.x;   // row over B*N
  if (i >= BB * NN) return;
  int b = i / NN, n = i % NN;
  const float* row = A + (long)b * NN * NN + (long)n * NN;
  float s = 0.f;
  for (int j = 0; j < NN; ++j) s += row[j];
  deg[i] = s + 1.f;                                 // rowsum(M) = rowsum(A)+1
  float nd = s - row[n];                            // off-diagonal degree
  inv_nd[i] = nd > 0.f ? 1.f / nd : 0.f;
}

// ------------------------------------------------------- WMMA f16 GEMM
// C[M_,N_] = A[M_,K_] * B^T  where Bt is stored row-major [N_,K_].
// Both fragments contiguous in K -> pure global_load_b128 feeding v_wmma.
// One wave computes a 16x64 C tile; all 4 B fragments are kept live so the
// 10 loads per K-step issue as clauses before the 4 WMMAs (partial waits).
// Requires K_ % 64 == 0 (true for all call sites: 512, 128, 64).
__global__ void k_gemm_wmma(const half_t* __restrict__ Am, const half_t* __restrict__ Bt,
                            float* __restrict__ Cm,
                            int N_, int K_, int lda, int ldb, int ldc,
                            long sA, long sB, long sC, int tiles) {
  int t = blockIdx.x * (blockDim.x >> 5) + (threadIdx.x >> 5);
  if (t >= tiles) return;
  int lane  = threadIdx.x & 31;
  int nloc  = lane & 15;
  int khalf = (lane >> 4) << 3;                     // 0 or 8
  int tilesN = N_ >> 6;
  int row0 = (t / tilesN) << 4;
  int col0 = (t % tilesN) << 6;
  const half_t* Ab = Am + (long)blockIdx.z * sA + (long)(row0 + nloc) * lda + khalf;
  const half_t* Bb = Bt + (long)blockIdx.z * sB + (long)(col0 + nloc) * ldb + khalf;
  float*        Cb = Cm + (long)blockIdx.z * sC;

  v8f acc[4];
#pragma unroll
  for (int j = 0; j < 4; ++j) acc[j] = (v8f){0.f, 0.f, 0.f, 0.f, 0.f, 0.f, 0.f, 0.f};

#pragma unroll 2
  for (int k0 = 0; k0 < K_; k0 += 32) {
    __builtin_prefetch(Ab + k0 + 256, 0, 1);        // global_prefetch_b8
    v8h a0 = *(const v8h*)(Ab + k0);
    v8h a1 = *(const v8h*)(Ab + k0 + 16);
    v8h b0[4], b1[4];
#pragma unroll
    for (int j = 0; j < 4; ++j) {                   // issue all B loads first
      const half_t* Bj = Bb + (long)(j << 4) * ldb;
      b0[j] = *(const v8h*)(Bj + k0);
      b1[j] = *(const v8h*)(Bj + k0 + 16);
    }
    v16h af = __builtin_shufflevector(a0, a1, 0, 1, 2, 3, 4, 5, 6, 7,
                                      8, 9, 10, 11, 12, 13, 14, 15);
#pragma unroll
    for (int j = 0; j < 4; ++j) {
      v16h bf = __builtin_shufflevector(b0[j], b1[j], 0, 1, 2, 3, 4, 5, 6, 7,
                                        8, 9, 10, 11, 12, 13, 14, 15);
      acc[j] = __builtin_amdgcn_wmma_f32_16x16x32_f16(false, af, false, bf,
                                                      (short)0, acc[j], false, false);
    }
  }
#pragma unroll
  for (int j = 0; j < 4; ++j) {
#pragma unroll
    for (int r = 0; r < 8; ++r) {
      int cm = r + khalf;                           // C/D rows: +8 for hi lane half
      Cb[(long)(row0 + cm) * ldc + col0 + (j << 4) + nloc] = acc[j][r];
    }
  }
}

// ---------------------------------------------------------- GIN #1 pieces
__global__ void k_hpre(const float* __restrict__ x, const float* __restrict__ eps,
                       float* __restrict__ hpre, half_t* __restrict__ hpreh, long n) {
  long i = (long)blockIdx.x * blockDim.x + threadIdx.x;
  if (i >= n) return;
  float v = (1.f + *eps) * x[i] + hpre[i];          // hpre held agg = A^T x
  hpre[i]  = v;
  hpreh[i] = (half_t)v;
}

__global__ void k_bn_stats(const float* __restrict__ Z, float* __restrict__ mean,
                           float* __restrict__ var, int rows) {
  int f = blockIdx.x, tid = threadIdx.x;
  __shared__ float ss[256], sq[256];
  float s = 0.f, q = 0.f;
  for (int r = tid; r < rows; r += 256) {
    float v = Z[(long)r * FF + f];
    s += v; q += v * v;
  }
  ss[tid] = s; sq[tid] = q;
  __syncthreads();
  for (int st = 128; st > 0; st >>= 1) {
    if (tid < st) { ss[tid] += ss[tid + st]; sq[tid] += sq[tid + st]; }
    __syncthreads();
  }
  if (tid == 0) {
    float mu = ss[0] / rows;
    mean[f] = mu;
    var[f]  = sq[0] / rows - mu * mu;               // biased variance
  }
}

__global__ void k_bn_apply(const float* __restrict__ Z, const float* __restrict__ mean,
                           const float* __restrict__ var, const float* __restrict__ g,
                           const float* __restrict__ be, float* __restrict__ out,
                           half_t* __restrict__ outh, long n) {
  long i = (long)blockIdx.x * blockDim.x + threadIdx.x;
  if (i >= n) return;
  int f = (int)(i % FF);
  float v = (Z[i] - mean[f]) * rsqrtf(var[f] + BN_EPS) * g[f] + be[f];
  v = v > 0.f ? v : 0.f;
  out[i] = v;
  if (outh) outh[i] = (half_t)v;
}

// ------------------------------------------------- Laplacian leader scoring
__global__ void k_row_norm(const float* __restrict__ h, float* __restrict__ nrm, int rows) {
  int i = blockIdx.x * blockDim.x + threadIdx.x;
  if (i >= rows) return;
  const float* r = h + (long)i * FF;
  float s = 0.f;
  for (int f = 0; f < FF; ++f) s += r[f] * r[f];
  nrm[i] = sqrtf(s);
}

__global__ void k_lap(const float* __restrict__ h, const float* __restrict__ P,
                      const float* __restrict__ deg, float* __restrict__ lap) {
  int i = blockIdx.x * blockDim.x + threadIdx.x;
  if (i >= BB * NN) return;
  const float* hr = h + (long)i * FF;
  const float* pr = P + (long)i * FF;
  float d = deg[i], s = 0.f;
  for (int f = 0; f < FF; ++f) { float v = d * hr[f] - pr[f]; s += v * v; }
  lap[i] = sqrtf(s);
}

__global__ void k_nei(const float* __restrict__ A, const float* __restrict__ lap,
                      const float* __restrict__ inv_nd, float* __restrict__ nei) {
  int i = blockIdx.x * blockDim.x + threadIdx.x;
  if (i >= BB * NN) return;
  int b = i / NN, n = i % NN;
  const float* row = A + (long)b * NN * NN + (long)n * NN;
  const float* lp  = lap + (long)b * NN;
  float dot = 0.f;
  for (int j = 0; j < NN; ++j) dot += row[j] * lp[j];
  dot -= row[n] * lp[n];                            // A_nd zeroes the diagonal
  nei[i] = lp[n] - inv_nd[i] * dot;
}

// top-K with lax.top_k tie-breaking (lower index wins on equal score)
__global__ void k_topk(const float* __restrict__ nei, int* __restrict__ leader) {
  int b = blockIdx.x, tid = threadIdx.x;            // 512 threads
  __shared__ float sval[NN];
  __shared__ float rv[NN];
  __shared__ int   ri[NN];
  float v = nei[(long)b * NN + tid];
  rv[tid] = v;
  __syncthreads();
  for (int s = 256; s > 0; s >>= 1) {
    if (tid < s) rv[tid] = fminf(rv[tid], rv[tid + s]);
    __syncthreads();
  }
  float mn = rv[0];
  __syncthreads();
  sval[tid] = v - mn + fabsf(mn);
  __syncthreads();
  for (int kk = 0; kk < KK; ++kk) {
    rv[tid] = sval[tid]; ri[tid] = tid;
    __syncthreads();
    for (int s = 256; s > 0; s >>= 1) {
      if (tid < s) {
        bool better = (rv[tid + s] > rv[tid]) ||
                      (rv[tid + s] == rv[tid] && ri[tid + s] < ri[tid]);
        if (better) { rv[tid] = rv[tid + s]; ri[tid] = ri[tid + s]; }
      }
      __syncthreads();
    }
    int win = ri[0];
    if (tid == 0) leader[b * KK + kk] = win;
    __syncthreads();
    if (tid == win) sval[tid] = -3.4e38f;
    __syncthreads();
  }
}

// ------------------------------------------------------- gathers / attention
__global__ void k_gather_clusters(const float* __restrict__ h, const int* __restrict__ leader,
                                  half_t* __restrict__ ch, const float* __restrict__ hn,
                                  float* __restrict__ cn) {
  long i = (long)blockIdx.x * blockDim.x + threadIdx.x;
  if (i >= (long)BB * KK * FF) return;
  int f = (int)(i % FF);
  long bk = i / FF;
  int b = (int)(bk / KK), k = (int)(bk % KK);
  int l = leader[b * KK + k];
  ch[i] = (half_t)h[((long)b * NN + l) * FF + f];
  if (f == 0) cn[bk] = hn[b * NN + l];
}

// McT[b][k][n] = M[n, leader_k]   (leader columns of M = A+I, transposed layout)
__global__ void k_gather_McT(const float* __restrict__ A, const int* __restrict__ leader,
                             half_t* __restrict__ McT) {
  long i = (long)blockIdx.x * blockDim.x + threadIdx.x;
  if (i >= (long)BB * KK * NN) return;
  int n = (int)(i % NN);
  long bk = i / NN;
  int b = (int)(bk / KK), k = (int)(bk % KK);
  int l = leader[b * KK + k];
  float v = A[(long)b * NN * NN + (long)n * NN + l] + (n == l ? 1.f : 0.f);
  McT[i] = (half_t)v;
}

__global__ void k_attn_fin(float* __restrict__ attn, const float* __restrict__ hn,
                           const float* __restrict__ cn, const half_t* __restrict__ McT,
                           const float* __restrict__ MMc) {
  long i = (long)blockIdx.x * blockDim.x + threadIdx.x;   // [B][N][K]
  if (i >= (long)BB * NN * KK) return;
  int k = (int)(i % KK);
  long bn = i / KK;
  int b = (int)(bn / NN), n = (int)(bn % NN);
  float a = attn[i] / (hn[bn] * cn[b * KK + k] + EPSC);
  float mval = (float)McT[((long)b * KK + k) * NN + n];
  float gs = fminf(mval, 1.f) + fminf(MMc[i], 1.f) + EPSC;  // G at leader columns
  attn[i] = a * gs / (1.f - SIGMA);
}

// per-row sparsemax over K=64; writes S^T in f16 directly ([B][K][N])
__global__ void k_sparsemax(const float* __restrict__ Zm, half_t* __restrict__ ShT) {
  int r = blockIdx.x * blockDim.x + threadIdx.x;
  if (r >= BB * NN) return;
  int b = r / NN, n = r % NN;
  const float* z = Zm + (long)r * KK;
  float zs[KK];
  for (int j = 0; j < KK; ++j) {                    // insertion sort, descending
    float v = z[j];
    int pos = j;
    while (pos > 0 && zs[pos - 1] < v) { zs[pos] = zs[pos - 1]; --pos; }
    zs[pos] = v;
  }
  float csum = 0.f, tnum = zs[0];
  int ksz = 1;
  for (int j = 0; j < KK; ++j) {
    csum += zs[j];
    if (1.f + (float)(j + 1) * zs[j] > csum) { ksz = j + 1; tnum = csum; }
  }
  float tau = (tnum - 1.f) / (float)ksz;
  for (int j = 0; j < KK; ++j) {
    float s = z[j] - tau;
    s = s > 0.f ? s : 0.f;
    ShT[((long)b * KK + j) * NN + n] = (half_t)s;
  }
}

// ------------------------------------------------------------ pooled graph
__global__ void k_apool_diag(float* __restrict__ Ap) {
  int i = blockIdx.x * blockDim.x + threadIdx.x;    // B*K
  if (i >= BB * KK) return;
  int b = i / KK, k = i % KK;
  Ap[(long)b * KK * KK + (long)k * KK + k] = 0.f;
}

__global__ void k_agg2(const float* __restrict__ Ap, const float* __restrict__ cf,
                       float* __restrict__ agg2) {
  long i = (long)blockIdx.x * blockDim.x + threadIdx.x;
  if (i >= (long)BB * KK * FF) return;
  int f = (int)(i % FF);
  long bl = i / FF;
  int b = (int)(bl / KK), l = (int)(bl % KK);
  const float* apb = Ap + (long)b * KK * KK;
  const float* cfb = cf + (long)b * KK * FF;
  float s = 0.f;
  for (int k2 = 0; k2 < KK; ++k2)
    if (apb[k2 * KK + l] != 0.f) s += cfb[k2 * FF + f];    // sum over sources
  agg2[i] = s;
}

__global__ void k_z2in(const float* __restrict__ cf, const float* __restrict__ agg2,
                       const float* __restrict__ eps, half_t* __restrict__ z2in, long n) {
  long i = (long)blockIdx.x * blockDim.x + threadIdx.x;
  if (i < n) z2in[i] = (half_t)((1.f + *eps) * cf[i] + agg2[i]);
}

__global__ void k_batch_out(float* __restrict__ out) {
  int i = blockIdx.x * blockDim.x + threadIdx.x;
  if (i < BB * KK) out[i] = (float)(i / KK);
}

// ================================================================= host side
static inline long gsz(long n, int b) { return (n + b - 1) / b; }

static void launch_gemm(const half_t* A, const half_t* Bt, float* C,
                        int M_, int N_, int K_, int lda, int ldb, int ldc,
                        long sA, long sB, long sC, int batch, hipStream_t st) {
  int tiles = (M_ / 16) * (N_ / 64);
  dim3 grid((unsigned)gsz(tiles, 8), 1, (unsigned)batch);
  k_gemm_wmma<<<grid, 256, 0, st>>>(A, Bt, C, N_, K_, lda, ldb, ldc, sA, sB, sC, tiles);
}

extern "C" void kernel_launch(void* const* d_in, const int* in_sizes, int n_in,
                              void* d_out, int out_size, void* d_ws, size_t ws_size,
                              hipStream_t stream) {
  (void)n_in; (void)out_size; (void)ws_size;
  const float* x    = (const float*)d_in[0];
  const float* W1   = (const float*)d_in[1];
  const float* g1   = (const float*)d_in[3];
  const float* be1  = (const float*)d_in[4];
  const float* eps1 = (const float*)d_in[5];
  const float* W2   = (const float*)d_in[6];
  const float* g2   = (const float*)d_in[8];
  const float* be2  = (const float*)d_in[9];
  const float* eps2 = (const float*)d_in[10];
  const long long* ei = (const long long*)d_in[11];
  long E = (long)in_sizes[11] / 2;

  char* ws = (char*)d_ws;
  size_t off = 0;
  auto alloc = [&](size_t bytes) -> void* {
    off = (off + 255) & ~(size_t)255;
    void* p = ws + off;
    off += bytes;
    return p;
  };
  const long nBN = (long)BB * NN * NN;
  const long nXF = (long)BB * NN * FF;
  const long nNK = (long)BB * NN * KK;
  const long nKF = (long)BB * KK * FF;
  float*  A      = (float*)alloc(nBN * 4);
  half_t* Mh     = (half_t*)alloc(nBN * 2);   // M = A + I, row-major f16
  half_t* AhT    = (half_t*)alloc(nBN * 2);   // A^T row-major f16
  half_t* XhT    = (half_t*)alloc(nXF * 2);   // [B][F][N]
  half_t* W1T    = (half_t*)alloc(FF * FF * 2);
  half_t* W2T    = (half_t*)alloc(FF * FF * 2);
  float*  hpre   = (float*)alloc(nXF * 4);
  half_t* hpreh  = (half_t*)alloc(nXF * 2);
  float*  Z      = (float*)alloc(nXF * 4);
  float*  h      = (float*)alloc(nXF * 4);
  half_t* hh     = (half_t*)alloc(nXF * 2);   // [B][N][F]
  half_t* hhT    = (half_t*)alloc(nXF * 2);   // [B][F][N]
  float*  P      = (float*)alloc(nXF * 4);
  float*  mean1  = (float*)alloc(FF * 4);
  float*  var1   = (float*)alloc(FF * 4);
  float*  mean2  = (float*)alloc(FF * 4);
  float*  var2   = (float*)alloc(FF * 4);
  float*  deg    = (float*)alloc((long)BB * NN * 4);
  float*  inv_nd = (float*)alloc((long)BB * NN * 4);
  float*  lap    = (float*)alloc((long)BB * NN * 4);
  float*  nei    = (float*)alloc((long)BB * NN * 4);
  float*  hn     = (float*)alloc((long)BB * NN * 4);
  float*  cn     = (float*)alloc((long)BB * KK * 4);
  int*    leader = (int*)alloc((long)BB * KK * 4);
  half_t* ch     = (half_t*)alloc(nKF * 2);   // [B][K][F] (= B^T layout for attn)
  half_t* McT    = (half_t*)alloc(nNK * 2);   // [B][K][N]
  float*  MMc    = (float*)alloc(nNK * 4);    // [B][N][K]
  float*  attn   = (float*)alloc(nNK * 4);    // [B][N][K]
  half_t* ShT    = (half_t*)alloc(nNK * 2);   // [B][K][N]
  float*  T      = (float*)alloc(nNK * 4);    // [B][N][K]
  half_t* ThT    = (half_t*)alloc(nNK * 2);   // [B][K][N]
  float*  cf     = (float*)alloc(nKF * 4);
  float*  Apool  = (float*)alloc((long)BB * KK * KK * 4);
  float*  agg2   = (float*)alloc(nKF * 4);
  half_t* z2in   = (half_t*)alloc(nKF * 2);
  float*  Z2     = (float*)alloc(nKF * 4);

  float* out_feat  = (float*)d_out;                 // [B*K, F]
  float* out_apool = out_feat + (long)BB * KK * FF; // [B, K, K]
  float* out_batch = out_apool + (long)BB * KK * KK;

  // 1. dense adjacency via scatter-add; f16 operands (M = A+I, and A^T)
  k_zero_f32<<<gsz(nBN, 256), 256, 0, stream>>>(A, nBN);
  k_scatter_edges<<<gsz(E, 256), 256, 0, stream>>>(ei, A, E);
  k_a_to_m_f16<<<gsz(nBN, 256), 256, 0, stream>>>(A, Mh, nBN);
  k_transpose_to_f16<<<gsz(nBN, 256), 256, 0, stream>>>(A, AhT, NN, NN, nBN);
  k_transpose_to_f16<<<gsz(nXF, 256), 256, 0, stream>>>(x, XhT, NN, FF, nXF);
  k_transpose_to_f16<<<gsz(FF * FF, 256), 256, 0, stream>>>(W1, W1T, FF, FF, FF * FF);
  k_transpose_to_f16<<<gsz(FF * FF, 256), 256, 0, stream>>>(W2, W2T, FF, FF, FF * FF);
  k_deg_nd<<<gsz(BB * NN, 256), 256, 0, stream>>>(A, deg, inv_nd);

  // 2. GIN #1: agg = A^T X  (segment_sum as GEMM), Z1 = hpre @ W1, BN+ReLU
  launch_gemm(AhT, XhT, hpre, NN, FF, NN, NN, NN, FF,
              (long)NN * NN, (long)NN * FF, (long)NN * FF, BB, stream);
  k_hpre<<<gsz(nXF, 256), 256, 0, stream>>>(x, eps1, hpre, hpreh, nXF);
  launch_gemm(hpreh, W1T, Z, BB * NN, FF, FF, FF, FF, FF, 0, 0, 0, 1, stream);
  k_bn_stats<<<FF, 256, 0, stream>>>(Z, mean1, var1, BB * NN);
  k_bn_apply<<<gsz(nXF, 256), 256, 0, stream>>>(Z, mean1, var1, g1, be1, h, hh, nXF);
  k_transpose_to_f16<<<gsz(nXF, 256), 256, 0, stream>>>(h, hhT, NN, FF, nXF);
  k_row_norm<<<gsz(BB * NN, 256), 256, 0, stream>>>(h, hn, BB * NN);

  // 3. Laplacian scores: P = M h, lap, nei, top-K leaders
  launch_gemm(Mh, hhT, P, NN, FF, NN, NN, NN, FF,
              (long)NN * NN, (long)NN * FF, (long)NN * FF, BB, stream);
  k_lap<<<gsz(BB * NN, 256), 256, 0, stream>>>(h, P, deg, lap);
  k_nei<<<gsz(BB * NN, 256), 256, 0, stream>>>(A, lap, inv_nd, nei);
  k_topk<<<BB, NN, 0, stream>>>(nei, leader);

  // 4. clusters, cosine attention, hop-2 prior restricted to leader columns
  k_gather_clusters<<<gsz(nKF, 256), 256, 0, stream>>>(h, leader, ch, hn, cn);
  k_gather_McT<<<gsz(nNK, 256), 256, 0, stream>>>(A, leader, McT);
  launch_gemm(hh, ch, attn, NN, KK, FF, FF, FF, KK,
              (long)NN * FF, (long)KK * FF, (long)NN * KK, BB, stream);
  launch_gemm(Mh, McT, MMc, NN, KK, NN, NN, NN, KK,
              (long)NN * NN, (long)KK * NN, (long)NN * KK, BB, stream);
  k_attn_fin<<<gsz(nNK, 256), 256, 0, stream>>>(attn, hn, cn, McT, MMc);
  k_sparsemax<<<gsz(BB * NN, 256), 256, 0, stream>>>(attn, ShT);

  // 5. pooled adjacency A_pool = S^T M S and features cf = S^T h
  launch_gemm(Mh, ShT, T, NN, KK, NN, NN, NN, KK,
              (long)NN * NN, (long)KK * NN, (long)NN * KK, BB, stream);
  k_transpose_to_f16<<<gsz(nNK, 256), 256, 0, stream>>>(T, ThT, NN, KK, nNK);
  launch_gemm(ShT, ThT, Apool, KK, KK, NN, NN, NN, KK,
              (long)KK * NN, (long)KK * NN, (long)KK * KK, BB, stream);
  k_apool_diag<<<gsz(BB * KK, 256), 256, 0, stream>>>(Apool);
  launch_gemm(ShT, hhT, cf, KK, FF, NN, NN, NN, FF,
              (long)KK * NN, (long)NN * FF, (long)KK * FF, BB, stream);

  // 6. GIN #2 on pooled graph
  k_agg2<<<gsz(nKF, 256), 256, 0, stream>>>(Apool, cf, agg2);
  k_z2in<<<gsz(nKF, 256), 256, 0, stream>>>(cf, agg2, eps2, z2in, nKF);
  launch_gemm(z2in, W2T, Z2, BB * KK, FF, FF, FF, FF, FF, 0, 0, 0, 1, stream);
  k_bn_stats<<<FF, 256, 0, stream>>>(Z2, mean2, var2, BB * KK);
  k_bn_apply<<<gsz(nKF, 256), 256, 0, stream>>>(Z2, mean2, var2, g2, be2,
                                                out_feat, (half_t*)nullptr, nKF);

  // 7. outputs: A_pool + new_batch
  k_copy_f32<<<gsz((long)BB * KK * KK, 256), 256, 0, stream>>>(Apool, out_apool,
                                                               (long)BB * KK * KK);
  k_batch_out<<<gsz(BB * KK, 256), 256, 0, stream>>>(out_batch);
}
